// ResBlock_Subject_5652176962243
// MI455X (gfx1250) — matrix operand order repaired and checked by reading, not compiled, not run on hardware
//
#include <hip/hip_runtime.h>

// MI455X / gfx1250, wave32. Implicit-GEMM conv via V_WMMA_F32_16X16X32_BF16.
//
// Roofline: ~55 GFLOP total, ~270 MB min HBM traffic -> ~12us floor at
// 23.3 TB/s. bf16 WMMA (f32 accum) keeps compute under the memory floor;
// intermediate h kept bf16 in d_ws (halves intermediate traffic).
//
// conv2's h-tile staging uses GLOBAL_LOAD_ASYNC_TO_LDS_B128 (ASYNCcnt):
// builtin confirmed present; params are v4i pointers in AS1 (global) / AS3 (LDS).

typedef __attribute__((ext_vector_type(16))) __bf16 v16bf;
typedef __attribute__((ext_vector_type(8)))  __bf16 v8bf;
typedef __attribute__((ext_vector_type(8)))  float  v8f;

#define EPSV 1e-5f

#if __has_builtin(__builtin_amdgcn_global_load_async_to_lds_b128) && \
    __has_builtin(__builtin_amdgcn_s_wait_asynccnt)
#define HAVE_ASYNC_LDS 1
typedef int v4i __attribute__((vector_size(16)));
typedef __attribute__((address_space(1))) v4i* gas_ptr;   // global v4i*
typedef __attribute__((address_space(3))) v4i* las_ptr;   // LDS v4i*
#else
#define HAVE_ASYNC_LDS 0
#endif

// ---------------- kernel 0: weight relayout f32 -> bf16 [s][k][co][ci] ----
__global__ void __launch_bounds__(256)
wcvt_kernel(const float* __restrict__ W1, const float* __restrict__ W2,
            __bf16* __restrict__ W1b, __bf16* __restrict__ W2b) {
  int i = blockIdx.x * 256 + threadIdx.x;
  if (i < 8 * 17 * 64 * 32) {            // W1: [8][64][32][17] -> [s][k][co][ci]
    int ci = i & 31; int j = i >> 5;
    int co = j & 63; j >>= 6;
    int k = j % 17, s = j / 17;
    W1b[i] = (__bf16)W1[((s * 64 + co) * 32 + ci) * 17 + k];
  }
  if (i < 8 * 17 * 64 * 64) {            // W2: [8][64][64][17] -> [s][k][co][ci]
    int ci = i & 63; int j = i >> 6;
    int co = j & 63; j >>= 6;
    int k = j % 17, s = j / 17;
    W2b[i] = (__bf16)W2[((s * 64 + co) * 64 + ci) * 17 + k];
  }
}

// ---------------- kernel 1: skip path: maxpool(4) + 1x1 conv -> y_out -----
__global__ void __launch_bounds__(256)
skip_kernel(const float* __restrict__ y, const int* __restrict__ subj,
            const float* __restrict__ Wskip, float* __restrict__ y_out) {
  __shared__ float yp[32][128];
  __shared__ float wk[64][32];
  const int b       = blockIdx.x >> 4;          // 16 tiles of 128 over 2048
  const int lo_base = (blockIdx.x & 15) * 128;
  const int t = threadIdx.x;
  const int s = subj[b];
  for (int i = t; i < 64 * 32; i += 256)
    wk[i >> 5][i & 31] = Wskip[s * 2048 + i];
  for (int e = t; e < 32 * 128; e += 256) {
    int ci = e >> 7, lp = e & 127;
    const float* ypt = y + ((size_t)b * 32 + ci) * 8192 + 4 * (lo_base + lp);
    yp[ci][lp] = fmaxf(fmaxf(ypt[0], ypt[1]), fmaxf(ypt[2], ypt[3]));
  }
  __syncthreads();
  for (int o = t; o < 64 * 128; o += 256) {
    int co = o >> 7, lp = o & 127;
    float sum = 0.0f;
#pragma unroll
    for (int ci = 0; ci < 32; ++ci) sum += wk[co][ci] * yp[ci][lp];
    y_out[((size_t)b * 64 + co) * 2048 + lo_base + lp] = sum;
  }
}

// ---------------- kernel 2: conv1 + BN1 + ReLU -> h (bf16 [b][l][ci]) -----
__global__ void __launch_bounds__(256)
conv1_kernel(const float* __restrict__ x, const int* __restrict__ subj,
             const __bf16* __restrict__ W1b,
             const float* __restrict__ g1, const float* __restrict__ b1,
             const float* __restrict__ m1, const float* __restrict__ v1,
             __bf16* __restrict__ h) {
  __shared__ __align__(32) __bf16 xs[144][32];   // 128 l + 2*8 halo, [l][ci]
  __shared__ float sa[64], sbi[64];
  const int b     = blockIdx.x >> 6;             // 64 l-tiles of 128
  const int lbase = (blockIdx.x & 63) * 128;
  const int t = threadIdx.x;
  const int s = subj[b];
  if (t < 64) {
    float a = g1[s * 64 + t] * rsqrtf(v1[s * 64 + t] + EPSV);
    sa[t]  = a;
    sbi[t] = b1[s * 64 + t] - m1[s * 64 + t] * a;
  }
  {
    // transpose + f32->bf16 convert: must go through VGPRs (no async path)
    const int ci = t >> 3;
    const int l0 = t & 7;
    const float* xp = x + ((size_t)b * 32 + ci) * 8192;
#pragma unroll
    for (int i = 0; i < 18; ++i) {
      int lpos = l0 + i * 8;
      int gl = lbase - 8 + lpos;
      float val = (gl >= 0 && gl < 8192) ? xp[gl] : 0.0f;
      xs[lpos][ci] = (__bf16)val;
    }
  }
  __syncthreads();

  const int lane = t & 31, wave = t >> 5;
  const int hv = lane >> 4, n = lane & 15;       // ISA half-wave / row-in-tile
  const int co_tile = wave & 3;                  // 4 tiles of 16 channels
  const int ngrp = wave >> 2;                    // 2 groups of 64 positions

  v8f acc[4];
#pragma unroll
  for (int i = 0; i < 4; ++i)
#pragma unroll
    for (int r = 0; r < 8; ++r) acc[i][r] = 0.0f;

  const __bf16* wbase = W1b + ((size_t)s * 17 * 64 + co_tile * 16 + n) * 32;
  for (int k = 0; k < 17; ++k) {                 // 17 taps x 1 K-slab of 32
    const __bf16* wp = wbase + (size_t)k * 64 * 32;
    // A fragment: lanes0-15 K=0-15 ; lanes16-31 K=16-31
    v8bf r1 = *(const v8bf*)(wp + 16 * hv);
    v8bf r2 = *(const v8bf*)(wp + 8 + 16 * hv);
    v16bf afrag = __builtin_shufflevector(r1, r2, 0, 1, 2, 3, 4, 5, 6, 7,
                                          8, 9, 10, 11, 12, 13, 14, 15);
#pragma unroll
    for (int nt = 0; nt < 4; ++nt) {
      int local = ngrp * 64 + nt * 16 + n + k;   // (l - lbase + 8) position
      // B fragment: element e -> K = 16*hv + e (contiguous ci in LDS)
      v16bf bfrag = *(const v16bf*)(&xs[local][16 * hv]);
      acc[nt] = __builtin_amdgcn_wmma_f32_16x16x32_bf16(
          false, afrag, false, bfrag, (short)0, acc[nt], false, false);
    }
  }

  const int cobase = co_tile * 16 + 8 * hv;      // D: M = r + 8*hv
#pragma unroll
  for (int nt = 0; nt < 4; ++nt) {
    int lcol = lbase + ngrp * 64 + nt * 16 + n;
    v8bf outv;
#pragma unroll
    for (int r = 0; r < 8; ++r) {
      int c = cobase + r;
      float val = fmaxf(acc[nt][r] * sa[c] + sbi[c], 0.0f);
      outv[r] = (__bf16)val;
    }
    *(v8bf*)(h + ((size_t)b * 8192 + lcol) * 64 + cobase) = outv;
  }
}

// ---------------- kernel 3: conv2(stride4) + ysk + BN2/ReLU -> outputs ----
__global__ void __launch_bounds__(256)
conv2_kernel(const __bf16* __restrict__ h, const int* __restrict__ subj,
             const __bf16* __restrict__ W2b,
             const float* __restrict__ g2, const float* __restrict__ b2,
             const float* __restrict__ m2, const float* __restrict__ v2,
             float* __restrict__ x_out, float* __restrict__ y_out) {
  __shared__ __align__(32) __bf16 hs[272][64];   // l in [4*lo_base-7, +269)
  __shared__ float sa[64], sbi[64];
  const int b       = blockIdx.x >> 5;           // 32 lo-tiles of 64
  const int lo_base = (blockIdx.x & 31) * 64;
  const int l_start = 4 * lo_base - 7;
  const int t = threadIdx.x;
  const int s = subj[b];
  if (t < 64) {
    float a = g2[s * 64 + t] * rsqrtf(v2[s * 64 + t] + EPSV);
    sa[t]  = a;
    sbi[t] = b2[s * 64 + t] - m2[s * 64 + t] * a;
  }
  // stage h tile: contiguous bf16 copy, 16-byte chunks (272 rows x 128B)
  for (int c = t; c < 272 * 8; c += 256) {
    int lpos = c >> 3, part = c & 7;
    int gl = l_start + lpos;
    if (gl >= 0 && gl < 8192) {
#if HAVE_ASYNC_LDS
      __builtin_amdgcn_global_load_async_to_lds_b128(
          (gas_ptr)(h + ((size_t)b * 8192 + gl) * 64 + part * 8),
          (las_ptr)(&hs[lpos][part * 8]), 0, 0);
#else
      *(v8bf*)(&hs[lpos][part * 8]) =
          *(const v8bf*)(h + ((size_t)b * 8192 + gl) * 64 + part * 8);
#endif
    } else {
      v8bf z;
#pragma unroll
      for (int i = 0; i < 8; ++i) z[i] = (__bf16)0.0f;
      *(v8bf*)(&hs[lpos][part * 8]) = z;
    }
  }
  // prefetch the ysk tile (y_out region) this block will read-modify-write
  {
    int co = t >> 2, seg = (t & 3) * 16;
    __builtin_prefetch(&y_out[((size_t)b * 64 + co) * 2048 + lo_base + seg], 0, 3);
  }
#if HAVE_ASYNC_LDS
  __builtin_amdgcn_s_wait_asynccnt(0);
#endif
  __syncthreads();

  const int lane = t & 31, wave = t >> 5;
  const int hv = lane >> 4, n = lane & 15;
  const int co_tile = wave & 3;
  const int ngrp = wave >> 2;                    // 2 groups of 32 positions

  v8f acc[2];
#pragma unroll
  for (int i = 0; i < 2; ++i)
#pragma unroll
    for (int r = 0; r < 8; ++r) acc[i][r] = 0.0f;

  const __bf16* wbase = W2b + ((size_t)s * 17 * 64 + co_tile * 16 + n) * 64;
  for (int k = 0; k < 17; ++k) {                 // 17 taps x 2 K-slabs of 32
    const __bf16* wkp = wbase + (size_t)k * 64 * 64;
#pragma unroll
    for (int slab = 0; slab < 2; ++slab) {
      const __bf16* wp = wkp + slab * 32;
      v8bf r1 = *(const v8bf*)(wp + 16 * hv);
      v8bf r2 = *(const v8bf*)(wp + 8 + 16 * hv);
      v16bf afrag = __builtin_shufflevector(r1, r2, 0, 1, 2, 3, 4, 5, 6, 7,
                                            8, 9, 10, 11, 12, 13, 14, 15);
#pragma unroll
      for (int nt = 0; nt < 2; ++nt) {
        int out_off = ngrp * 32 + nt * 16 + n;   // lo - lo_base in [0,64)
        int local = 4 * out_off + k;             // l - l_start in [0,269)
        v16bf bfrag = *(const v16bf*)(&hs[local][slab * 32 + 16 * hv]);
        acc[nt] = __builtin_amdgcn_wmma_f32_16x16x32_bf16(
            false, afrag, false, bfrag, (short)0, acc[nt], false, false);
      }
    }
  }

  const int cobase = co_tile * 16 + 8 * hv;
#pragma unroll
  for (int nt = 0; nt < 2; ++nt) {
    int lo = lo_base + ngrp * 32 + nt * 16 + n;
#pragma unroll
    for (int r = 0; r < 8; ++r) {
      int c = cobase + r;
      size_t idx = ((size_t)b * 64 + c) * 2048 + lo;
      float x2 = acc[nt][r] + y_out[idx];        // add precomputed ysk
      y_out[idx] = x2;                            // y_out = x2
      x_out[idx] = fmaxf(x2 * sa[c] + sbi[c], 0.0f);
    }
  }
}

extern "C" void kernel_launch(void* const* d_in, const int* in_sizes, int n_in,
                              void* d_out, int out_size, void* d_ws, size_t ws_size,
                              hipStream_t stream) {
  const float* x     = (const float*)d_in[0];
  const float* y     = (const float*)d_in[1];
  const int*   subj  = (const int*)d_in[2];
  const float* W1    = (const float*)d_in[3];
  const float* g1    = (const float*)d_in[4];
  const float* b1    = (const float*)d_in[5];
  const float* m1    = (const float*)d_in[6];
  const float* v1    = (const float*)d_in[7];
  const float* W2    = (const float*)d_in[8];
  const float* g2    = (const float*)d_in[9];
  const float* b2    = (const float*)d_in[10];
  const float* m2    = (const float*)d_in[11];
  const float* v2    = (const float*)d_in[12];
  const float* Wskip = (const float*)d_in[13];

  float* x_out = (float*)d_out;
  float* y_out = x_out + (size_t)64 * 64 * 2048;

  // workspace layout (all 256B-aligned): W1b | W2b | h(bf16)
  __bf16* W1b  = (__bf16*)d_ws;                                   // 557,056 B
  __bf16* W2b  = (__bf16*)((char*)d_ws + 557056);                 // 1,114,112 B
  __bf16* hbuf = (__bf16*)((char*)d_ws + 1671168);                // 67,108,864 B

  wcvt_kernel <<<2176, 256, 0, stream>>>(W1, W2, W1b, W2b);
  skip_kernel <<<64 * 16, 256, 0, stream>>>(y, subj, Wskip, y_out);
  conv1_kernel<<<64 * 64, 256, 0, stream>>>(x, subj, W1b, g1, b1, m1, v1, hbuf);
  conv2_kernel<<<64 * 32, 256, 0, stream>>>(hbuf, subj, W2b, g2, b2, m2, v2,
                                            x_out, y_out);
}
